// Sender_10462540333588
// MI455X (gfx1250) — compile-verified, matrix operand order checked by reading, not run on hardware
//
#include <hip/hip_runtime.h>
#include <hip/hip_bf16.h>

// MI455X (gfx1250) LSTM greedy decoder.
// Dominant cost: 32 sequential [128,512]x[512,32000] vocab GEMMs (~134 GFLOP).
// lp_W in bf16 = 32.8 MB -> resident in 192MB L2 -> compute bound -> use
// v_wmma_f32_16x16x32_bf16 with f32 accumulation for every GEMM.
// Global->LDS staging uses gfx1250 async-to-LDS loads (ASYNCcnt) with
// double-buffered LDS so the next K-chunk streams in under the WMMAs.

typedef __attribute__((ext_vector_type(16))) __bf16 v16bf;
typedef __attribute__((ext_vector_type(8)))  __bf16 v8bf;
typedef __attribute__((ext_vector_type(8)))  float  v8f;

// Pointer types for the async-to-LDS builtin (param is int __vector(4) ptr).
typedef int b128vec __attribute__((vector_size(16)));
typedef __attribute__((address_space(1))) b128vec* gptr128;
typedef __attribute__((address_space(3))) b128vec* lptr128;

#define B_   128
#define F_   2048
#define V_   32000
#define E_   256
#define H_   512
#define T_   32
#define G4H  2048   // 4*H
#define KC   64     // K chunk staged in LDS
#define NT   128    // N tile per workgroup
#define CHUNK_ELEMS (128 * KC)

#if __has_builtin(__builtin_amdgcn_global_load_async_to_lds_b128) && \
    __has_builtin(__builtin_amdgcn_s_wait_asynccnt)
#define HAVE_ASYNC_LDS 1
#else
#define HAVE_ASYNC_LDS 0
#endif

// ---- fragment loader (ISA 7.12.2 16-bit A layout: lane<16 -> K 0..7|16..23,
// lane>=16 -> K 8..15|24..31; row = lane&15) ----
__device__ __forceinline__ v16bf load_frag(const __bf16* base, int ld, int lane) {
  const int r = lane & 15;
  const int hsel = lane >> 4;
  const __bf16* p = base + r * ld + 8 * hsel;
  v8bf lo = *(const v8bf*)(p);
  v8bf hi = *(const v8bf*)(p + 16);
  v16bf f;
#pragma unroll
  for (int i = 0; i < 8; ++i) { f[i] = lo[i]; f[i + 8] = hi[i]; }
  return f;
}

// Stage one 128xKC chunk of A and of W into LDS (16 KB each).
__device__ __forceinline__ void stage_chunk(const __bf16* __restrict__ A,
                                            const __bf16* __restrict__ W,
                                            int K, int n0, int k0,
                                            __bf16* sA, __bf16* sW, int tid) {
#pragma unroll
  for (int i = 0; i < 4; ++i) {
    const int e = (tid + i * 256) * 8;      // flat bf16 element index
    const int r = e >> 6;                   // row (ld = KC = 64)
    const int c = e & 63;
#if HAVE_ASYNC_LDS
    // gfx1250 GLOBAL_LOAD_ASYNC_TO_LDS_B128 (ASYNCcnt) -- bypasses VGPRs
    __builtin_amdgcn_global_load_async_to_lds_b128(
        (gptr128)(A + (size_t)r * K + k0 + c), (lptr128)(sA + e), 0, 0);
    __builtin_amdgcn_global_load_async_to_lds_b128(
        (gptr128)(W + (size_t)(n0 + r) * K + k0 + c), (lptr128)(sW + e), 0, 0);
#else
    *(uint4*)(sA + e) = *(const uint4*)(A + (size_t)r * K + k0 + c);
    *(uint4*)(sW + e) = *(const uint4*)(W + (size_t)(n0 + r) * K + k0 + c);
#endif
  }
}

// One full K sweep of C[128 x NT] += A[128 x K] * W[n0..n0+NT, 0..K]^T
// sA/sW each hold 2 double-buffered 128xKC chunks.
__device__ __forceinline__ void gemm_pass(v8f acc[8],
                                          const __bf16* __restrict__ A,
                                          const __bf16* __restrict__ W,
                                          int K, int n0,
                                          __bf16* sA, __bf16* sW,
                                          int tid, int lane, int wave) {
  int buf = 0;
  stage_chunk(A, W, K, n0, 0, sA, sW, tid);
  for (int k0 = 0; k0 < K; k0 += KC) {
#if HAVE_ASYNC_LDS
    __builtin_amdgcn_s_wait_asynccnt(0);
#endif
    __syncthreads();
    if (k0 + KC < K)  // stream next chunk into the other buffer under the WMMAs
      stage_chunk(A, W, K, n0, k0 + KC,
                  sA + (buf ^ 1) * CHUNK_ELEMS, sW + (buf ^ 1) * CHUNK_ELEMS, tid);
    const __bf16* sAc = sA + buf * CHUNK_ELEMS;
    const __bf16* sWc = sW + buf * CHUNK_ELEMS;
#pragma unroll
    for (int kk = 0; kk < KC; kk += 32) {
      v16bf aF = load_frag(sAc + wave * 16 * KC + kk, KC, lane);
      v16bf bCur = load_frag(sWc + kk, KC, lane);
#pragma unroll
      for (int nt = 0; nt < 8; ++nt) {
        v16bf bNxt = bCur;
        if (nt < 7) bNxt = load_frag(sWc + (nt + 1) * 16 * KC + kk, KC, lane);
        acc[nt] = __builtin_amdgcn_wmma_f32_16x16x32_bf16(
            false, aF, false, bCur, (short)0, acc[nt], false, false);
        bCur = bNxt;
      }
    }
    buf ^= 1;
  }
  __syncthreads();  // protect LDS reuse by a following pass
}

// Generic fused GEMM: out[128, N] = A1*W1^T (+ A2*W2^T) + bias1 (+ bias2)
// grid.x = N / NT, block = 256 (8 waves). W row-major [N, K].
__global__ __launch_bounds__(256)
void wmma_gemm_kernel(const __bf16* __restrict__ A1, const __bf16* __restrict__ W1, int K1,
                      const __bf16* __restrict__ A2, const __bf16* __restrict__ W2, int K2,
                      const float* __restrict__ bias1, const float* __restrict__ bias2,
                      float* __restrict__ outF, __bf16* __restrict__ outB, int ldout) {
  __shared__ __attribute__((aligned(16))) __bf16 sA[2 * CHUNK_ELEMS];
  __shared__ __attribute__((aligned(16))) __bf16 sW[2 * CHUNK_ELEMS];
  const int tid  = threadIdx.x;
  const int lane = tid & 31;
  const int wave = tid >> 5;
  const int n0   = blockIdx.x * NT;

  v8f acc[8];
#pragma unroll
  for (int i = 0; i < 8; ++i)
#pragma unroll
    for (int j = 0; j < 8; ++j) acc[i][j] = 0.f;

  gemm_pass(acc, A1, W1, K1, n0, sA, sW, tid, lane, wave);
  if (A2) gemm_pass(acc, A2, W2, K2, n0, sA, sW, tid, lane, wave);

  // C/D layout (ISA 7.12.2): VGPR v, lanes 0-15 -> M=v, lanes 16-31 -> M=v+8; N=lane%16
  const int mbase = wave * 16 + 8 * (lane >> 4);
  const int nbase = n0 + (lane & 15);
#pragma unroll
  for (int nt = 0; nt < 8; ++nt) {
    const int n = nbase + nt * 16;
    const float bs = (bias1 ? bias1[n] : 0.f) + (bias2 ? bias2[n] : 0.f);
#pragma unroll
    for (int v = 0; v < 8; ++v) {
      const float val = acc[nt][v] + bs;
      const int m = mbase + v;
      if (outF) outF[(size_t)m * ldout + n] = val;
      if (outB) outB[(size_t)m * ldout + n] = (__bf16)val;
    }
  }
}

__global__ void f32_to_bf16(const float* __restrict__ x, __bf16* __restrict__ y, int n) {
  int i = blockIdx.x * blockDim.x + threadIdx.x;
  const int stride = gridDim.x * blockDim.x;
  for (; i < n; i += stride) y[i] = (__bf16)x[i];
}

__global__ void init_state(float* __restrict__ c, __bf16* __restrict__ w,
                           const __bf16* __restrict__ embb, const int* __restrict__ start) {
  const int i = blockIdx.x * 256 + threadIdx.x;   // 65536 total = B*H
  if (i < B_ * H_) c[i] = 0.f;
  if (i < B_ * E_) {
    const int e = i & (E_ - 1);
    w[i] = embb[(size_t)start[0] * E_ + e];
  }
}

__global__ __launch_bounds__(256)
void lstm_pointwise(const float* __restrict__ gates, float* __restrict__ c,
                    __bf16* __restrict__ hb) {
  const int i = blockIdx.x * 256 + threadIdx.x;   // 0 .. B*H-1
  const int b = i >> 9;
  const int j = i & (H_ - 1);
  const float* g = gates + (size_t)b * G4H;
  const float ig = g[j], fg = g[H_ + j], gg = g[2 * H_ + j], og = g[3 * H_ + j];
  const float si = 1.f / (1.f + __expf(-ig));
  const float sf = 1.f / (1.f + __expf(-fg));
  const float tg = tanhf(gg);
  const float so = 1.f / (1.f + __expf(-og));
  const float cn = sf * c[i] + si * tg;
  c[i] = cn;
  hb[i] = (__bf16)(so * tanhf(cn));
}

// One block per batch row: argmax + log-softmax(argmax) + embedding gather.
__global__ __launch_bounds__(256)
void argmax_pick(const float* __restrict__ logits, const __bf16* __restrict__ embb,
                 __bf16* __restrict__ w, float* __restrict__ out, int step) {
  __shared__ float smax[256];
  __shared__ int   sidx[256];
  __shared__ float ssum[256];
  __shared__ int   sgidx;
  __shared__ float sM;
  const int b = blockIdx.x;
  const int t = threadIdx.x;
  const float* row = logits + (size_t)b * V_;

  float m = -3.4e38f; int mi = 0;
  for (int cx = t; cx < V_; cx += 256) {
    const float x = row[cx];
    if (x > m) { m = x; mi = cx; }
  }
  smax[t] = m; sidx[t] = mi;
  __syncthreads();
  for (int s = 128; s > 0; s >>= 1) {
    if (t < s) {
      const float om = smax[t + s]; const int oi = sidx[t + s];
      if (om > smax[t] || (om == smax[t] && oi < sidx[t])) { smax[t] = om; sidx[t] = oi; }
    }
    __syncthreads();
  }
  if (t == 0) { sM = smax[0]; sgidx = sidx[0]; }
  __syncthreads();
  const float M = sM;
  float sl = 0.f;
  for (int cx = t; cx < V_; cx += 256) sl += __expf(row[cx] - M);
  ssum[t] = sl;
  __syncthreads();
  for (int s = 128; s > 0; s >>= 1) { if (t < s) ssum[t] += ssum[t + s]; __syncthreads(); }
  const int gidx = sgidx;
  if (t == 0) {
    out[(size_t)b * T_ + step] = (float)gidx;     // message[b, step]
    out[B_ * T_ + b] = -__logf(ssum[0]);          // logp of picked token (last step survives)
  }
  for (int e = t; e < E_; e += 256) w[(size_t)b * E_ + e] = embb[(size_t)gidx * E_ + e];
}

extern "C" void kernel_launch(void* const* d_in, const int* in_sizes, int n_in,
                              void* d_out, int out_size, void* d_ws, size_t ws_size,
                              hipStream_t stream) {
  (void)in_sizes; (void)n_in; (void)out_size; (void)ws_size;
  const float* t_in  = (const float*)d_in[0];
  const int*   start = (const int*)d_in[1];
  const float* aff_W = (const float*)d_in[3];
  const float* aff_b = (const float*)d_in[4];
  const float* emb   = (const float*)d_in[5];
  const float* W_ih  = (const float*)d_in[6];
  const float* W_hh  = (const float*)d_in[7];
  const float* b_ih  = (const float*)d_in[8];
  const float* b_hh  = (const float*)d_in[9];
  const float* lp_W  = (const float*)d_in[10];
  const float* lp_b  = (const float*)d_in[11];
  float* out = (float*)d_out;

  char* ws = (char*)d_ws;
  size_t off = 0;
  auto alloc = [&](size_t bytes) -> char* {
    char* p = ws + off;
    off = (off + bytes + 255) & ~(size_t)255;
    return p;
  };
  __bf16* t_b    = (__bf16*)alloc((size_t)B_ * F_ * 2);
  __bf16* affW_b = (__bf16*)alloc((size_t)H_ * F_ * 2);
  __bf16* emb_b  = (__bf16*)alloc((size_t)V_ * E_ * 2);
  __bf16* Wih_b  = (__bf16*)alloc((size_t)G4H * E_ * 2);
  __bf16* Whh_b  = (__bf16*)alloc((size_t)G4H * H_ * 2);
  __bf16* lpW_b  = (__bf16*)alloc((size_t)V_ * H_ * 2);
  __bf16* h_b    = (__bf16*)alloc((size_t)B_ * H_ * 2);
  float*  c_f    = (float*)alloc((size_t)B_ * H_ * 4);
  __bf16* w_b    = (__bf16*)alloc((size_t)B_ * E_ * 2);
  float*  gates  = (float*)alloc((size_t)B_ * G4H * 4);
  float*  logits = (float*)alloc((size_t)B_ * V_ * 4);

  auto cgrid = [](long n) { long g = (n + 255) / 256; return (int)(g > 2048 ? 2048 : g); };
  f32_to_bf16<<<cgrid((long)B_ * F_), 256, 0, stream>>>(t_in, t_b, B_ * F_);
  f32_to_bf16<<<cgrid((long)H_ * F_), 256, 0, stream>>>(aff_W, affW_b, H_ * F_);
  f32_to_bf16<<<cgrid((long)V_ * E_), 256, 0, stream>>>(emb, emb_b, V_ * E_);
  f32_to_bf16<<<cgrid((long)G4H * E_), 256, 0, stream>>>(W_ih, Wih_b, G4H * E_);
  f32_to_bf16<<<cgrid((long)G4H * H_), 256, 0, stream>>>(W_hh, Whh_b, G4H * H_);
  f32_to_bf16<<<cgrid((long)V_ * H_), 256, 0, stream>>>(lp_W, lpW_b, V_ * H_);

  // h0 = t @ aff_W^T + aff_b  -> bf16 h
  wmma_gemm_kernel<<<H_ / NT, 256, 0, stream>>>(t_b, affW_b, F_, nullptr, nullptr, 0,
                                                aff_b, nullptr, nullptr, h_b, H_);
  init_state<<<(B_ * H_) / 256, 256, 0, stream>>>(c_f, w_b, emb_b, start);

  for (int stepi = 0; stepi < T_; ++stepi) {
    // gates = w @ W_ih^T + h @ W_hh^T + b_ih + b_hh
    wmma_gemm_kernel<<<G4H / NT, 256, 0, stream>>>(w_b, Wih_b, E_, h_b, Whh_b, H_,
                                                   b_ih, b_hh, gates, nullptr, G4H);
    lstm_pointwise<<<(B_ * H_) / 256, 256, 0, stream>>>(gates, c_f, h_b);
    // logits = h @ lp_W^T + lp_b   (dominant GEMM, lp_W resident in L2)
    wmma_gemm_kernel<<<V_ / NT, 256, 0, stream>>>(h_b, lpW_b, H_, nullptr, nullptr, 0,
                                                  lp_b, nullptr, logits, nullptr, V_);
    argmax_pick<<<B_, 256, 0, stream>>>(logits, emb_b, w_b, out, stepi);
  }
}
